// CEN_BRL_22763326668900
// MI455X (gfx1250) — compile-verified, matrix operand order checked by reading, not run on hardware
//
#include <hip/hip_runtime.h>
#include <hip/hip_bf16.h>
#include <math.h>

typedef __attribute__((ext_vector_type(16))) __bf16 v16bf;
typedef __attribute__((ext_vector_type(8)))  __bf16 bf16x8;
typedef __attribute__((ext_vector_type(8)))  float  v8f;

#define N_TRAIN 4096
#define N_FEAT  64
#define ENC     256
#define N_ANTES 2048
#define GATES   1024
#define MAX_LEN 16

__device__ __forceinline__ float sigf(float x){ return 1.0f/(1.0f+__expf(-x)); }

// ---- gfx1250 async copy: global -> LDS, 16B per lane, tracked by ASYNCcnt ----
__device__ __forceinline__ void async_load_b128(unsigned lds_off, const void* gaddr){
  asm volatile("global_load_async_to_lds_b128 %0, %1, off"
               :: "v"(lds_off), "v"((unsigned long long)(size_t)gaddr)
               : "memory");
}
__device__ __forceinline__ void wait_async0(){
  asm volatile("s_wait_asynccnt 0x0" ::: "memory");
}

// ---- WMMA fragment load: 16-bit operand, grouped-K layout (ISA 7.12.2) ----
__device__ __forceinline__ v16bf load_frag(const __bf16* __restrict__ base, int line,
                                           int k0, int ld, int hk){
  const __bf16* p = base + (size_t)line*ld + k0 + 8*hk;
  bf16x8 lo = *(const bf16x8*)p;
  bf16x8 hi = *(const bf16x8*)(p + 16);
  v16bf r;
#pragma unroll
  for(int i=0;i<8;i++){ r[i]=lo[i]; r[i+8]=hi[i]; }
  return r;
}

// ---- Generic C = A(MxK,bf16) * B(NxK,bf16 "weight rows")^T [+bias][relu], f32 out, optional bf16 copy
__global__ void k_wmma_gemm(const __bf16* __restrict__ A, const __bf16* __restrict__ B,
                            const float* __restrict__ bias, float* __restrict__ C,
                            __bf16* __restrict__ Cbf,
                            int M, int N, int K, int lda, int ldb, int ldc, int relu)
{
  int lane = threadIdx.x & 31;
  int wave = blockIdx.x * (blockDim.x >> 5) + (threadIdx.x >> 5);
  int tilesN = N >> 6;                       // 16x64 tile per wave
  int m0 = (wave / tilesN) << 4;
  int n0 = (wave % tilesN) << 6;
  if (m0 >= M) return;                       // wave-uniform exit (EXEC stays all-ones)
  int hk = lane >> 4, ln = lane & 15;

  v8f acc[4] = {};
  for (int kb = 0; kb < K; kb += 32) {
    v16bf a = load_frag(A, m0 + ln, kb, lda, hk);
#pragma unroll
    for (int t = 0; t < 4; t++) {
      v16bf b = load_frag(B, n0 + 16*t + ln, kb, ldb, hk);
      acc[t] = __builtin_amdgcn_wmma_f32_16x16x32_bf16(false, a, false, b,
                                                       (short)0, acc[t], false, false);
    }
  }
#pragma unroll
  for (int t = 0; t < 4; t++)
#pragma unroll
    for (int r = 0; r < 8; r++) {
      int row = m0 + r + 8*hk;
      int col = n0 + 16*t + ln;
      float v = acc[t][r];
      if (bias) v += bias[col];
      if (relu) v = v > 0.f ? v : 0.f;
      C[(size_t)row*ldc + col] = v;
      if (Cbf) Cbf[(size_t)row*ldc + col] = (__bf16)v;
    }
}

// ---- z = Hbf*w_hh^T + (b_ih+b_hh) + s_row * rowsum(w_ih);  s_row = step0?1:S[row,idx]
// Block = 128x64 output tile (8 waves x 16 rows); B tile (64x256 bf16 = 32KB) staged
// once per block into LDS via global_load_async_to_lds_b128 and shared by all waves.
__global__ void k_wmma_z(const __bf16* __restrict__ Hbf, const __bf16* __restrict__ Wbf,
                         const float* __restrict__ b_ih, const float* __restrict__ b_hh,
                         const float* __restrict__ sums, const float* __restrict__ S,
                         const int* __restrict__ idx, int step0, float* __restrict__ Z)
{
  extern __shared__ __bf16 Bs[];             // 64 rows x 256 k, row-major (32 KB)
  int lane = threadIdx.x & 31;
  int wave = threadIdx.x >> 5;               // 0..7
  int mb = blockIdx.x >> 4;                  // 0..31
  int nb = blockIdx.x & 15;                  // 0..15
  int m0 = mb*128 + wave*16;
  int n0 = nb*64;
  int hk = lane >> 4, ln = lane & 15;

  // ---- async stage of B tile: contiguous 32KB from Wbf + n0*ENC ----
  const char* gB = (const char*)(Wbf + (size_t)n0*ENC);
  unsigned tbase = threadIdx.x * 16;
#pragma unroll
  for (int i = 0; i < 8; i++) {
    unsigned off = tbase + i*4096;
    async_load_b128(off, gB + off);
  }
  wait_async0();
  __syncthreads();

  int sel = step0 ? 0 : *idx;
  float srow[8];
#pragma unroll
  for (int r = 0; r < 8; r++) {
    int row = m0 + r + 8*hk;
    srow[r] = step0 ? 1.0f : S[(size_t)row*N_ANTES + sel];
  }

  v8f acc[4] = {};
#pragma unroll
  for (int kb = 0; kb < ENC; kb += 32) {
    __builtin_prefetch(Hbf + (size_t)(m0 + ln)*ENC + kb + 32, 0, 1);
    v16bf a = load_frag(Hbf, m0 + ln, kb, ENC, hk);
#pragma unroll
    for (int t = 0; t < 4; t++) {
      v16bf b = load_frag(Bs, 16*t + ln, kb, ENC, hk);   // ds_load path
      acc[t] = __builtin_amdgcn_wmma_f32_16x16x32_bf16(false, a, false, b,
                                                       (short)0, acc[t], false, false);
    }
  }
#pragma unroll
  for (int t = 0; t < 4; t++)
#pragma unroll
    for (int r = 0; r < 8; r++) {
      int row = m0 + r + 8*hk;
      int col = n0 + 16*t + ln;
      Z[(size_t)row*GATES + col] = acc[t][r] + b_ih[col] + b_hh[col] + srow[r]*sums[col];
    }
}

// ---- small kernels ----
__global__ void k_cvt(const float* __restrict__ src, __bf16* __restrict__ dst,
                      int rows, int cols, int ldsrc){
  int i = blockIdx.x*blockDim.x + threadIdx.x;
  if (i >= rows*cols) return;
  int r = i / cols, cc = i % cols;
  dst[i] = (__bf16)src[(size_t)r*ldsrc + cc];
}

__global__ void k_transpose_cvt(const float* __restrict__ S, __bf16* __restrict__ St){
  int i = blockIdx.x*blockDim.x + threadIdx.x;   // over 4096*2048, S row-major
  int n = i / N_ANTES, a = i % N_ANTES;
  St[(size_t)a*N_TRAIN + n] = (__bf16)S[i];
}

__global__ void k_rowsum(const float* __restrict__ w_ih, float* __restrict__ sums){
  int j = blockIdx.x*blockDim.x + threadIdx.x;
  if (j >= GATES) return;
  const float* r = w_ih + (size_t)j*N_ANTES;
  float s = 0.f;
  for (int a = 0; a < N_ANTES; a++) s += r[a];
  sums[j] = s;
}

__global__ void k_zero(float* p, int n){
  int i = blockIdx.x*blockDim.x + threadIdx.x;
  if (i < n) p[i] = 0.f;
}

__global__ void k_gates(const float* __restrict__ z, float* __restrict__ c,
                        float* __restrict__ h, __bf16* __restrict__ hbf){
  int n = blockIdx.x, k = threadIdx.x;
  const float* zr = z + (size_t)n*GATES;
  float vi = sigf(zr[k]);
  float vf = sigf(zr[ENC + k]);
  float vg = tanhf(zr[2*ENC + k]);
  float vo = sigf(zr[3*ENC + k]);
  size_t o = (size_t)n*ENC + k;
  float cn = vf*c[o] + vi*vg;
  c[o] = cn;
  float hn = vo*tanhf(cn);
  h[o] = hn;
  hbf[o] = (__bf16)hn;
}

__global__ void k_col_mean(const float* __restrict__ h, float* __restrict__ e){
  int k = blockIdx.x;                        // 256 columns
  float s = 0.f;
  for (int n = threadIdx.x; n < N_TRAIN; n += 256) s += h[(size_t)n*ENC + k];
  __shared__ float sm[256];
  sm[threadIdx.x] = s; __syncthreads();
  for (int off = 128; off; off >>= 1){
    if (threadIdx.x < off) sm[threadIdx.x] += sm[threadIdx.x + off];
    __syncthreads();
  }
  if (threadIdx.x == 0) e[k] = sm[0] * (1.0f / N_TRAIN);
}

__global__ void k_ewb(const float* __restrict__ e, const float* __restrict__ att_w1,
                      const float* __restrict__ att_b1, float* __restrict__ ewb){
  int k = threadIdx.x;
  const float* wb = att_w1 + (size_t)k*(N_TRAIN + ENC) + N_TRAIN;
  float s = 0.f;
  for (int j = 0; j < ENC; j++) s += e[j]*wb[j];
  ewb[k] = s + att_b1[k];
}

__global__ void k_scores(const float* __restrict__ pre, const float* __restrict__ ewb,
                         const float* __restrict__ att_w2, const float* __restrict__ att_b2,
                         float* __restrict__ scores){
  int a = blockIdx.x*blockDim.x + threadIdx.x;
  const float* pr = pre + (size_t)a*ENC;
  float s = 0.f;
  for (int k = 0; k < ENC; k++){
    float v = pr[k] + ewb[k];
    v = v > 0.f ? v : 0.f;
    s += v*att_w2[k];
  }
  scores[a] = s + att_b2[0];
}

__global__ void k_softmax(const float* __restrict__ scores, float* __restrict__ logp,
                          int* __restrict__ idx){
  __shared__ float sv[1024]; __shared__ int si[1024];
  int t = threadIdx.x;
  float m = -1e30f; int mi = 0;
  for (int a = t; a < N_ANTES; a += 1024){
    float v = scores[a];
    if (v > m){ m = v; mi = a; }
  }
  sv[t] = m; si[t] = mi; __syncthreads();
  for (int off = 512; off; off >>= 1){
    if (t < off){
      if (sv[t+off] > sv[t] || (sv[t+off] == sv[t] && si[t+off] < si[t])){
        sv[t] = sv[t+off]; si[t] = si[t+off];
      }
    }
    __syncthreads();
  }
  float M = sv[0]; int best = si[0];
  __syncthreads();
  float se = 0.f;
  for (int a = t; a < N_ANTES; a += 1024) se += __expf(scores[a] - M);
  sv[t] = se; __syncthreads();
  for (int off = 512; off; off >>= 1){
    if (t < off) sv[t] += sv[t+off];
    __syncthreads();
  }
  float lse = M + __logf(sv[0]);
  for (int a = t; a < N_ANTES; a += 1024) logp[a] = scores[a] - lse;
  if (t == 0) *idx = best;
}

extern "C" void kernel_launch(void* const* d_in, const int* in_sizes, int n_in,
                              void* d_out, int out_size, void* d_ws, size_t ws_size,
                              hipStream_t stream) {
  (void)in_sizes; (void)n_in; (void)out_size; (void)ws_size;
  const float* context = (const float*)d_in[0];
  const float* S       = (const float*)d_in[1];
  const float* enc_w1  = (const float*)d_in[2];
  const float* enc_b1  = (const float*)d_in[3];
  const float* enc_w2  = (const float*)d_in[4];
  const float* enc_b2  = (const float*)d_in[5];
  const float* w_ih    = (const float*)d_in[6];
  const float* w_hh    = (const float*)d_in[7];
  const float* b_ih    = (const float*)d_in[8];
  const float* b_hh    = (const float*)d_in[9];
  const float* att_w1  = (const float*)d_in[10];
  const float* att_b1  = (const float*)d_in[11];
  const float* att_w2  = (const float*)d_in[12];
  const float* att_b2  = (const float*)d_in[13];
  float* out = (float*)d_out;

  char* w = (char*)d_ws;
  auto alloc = [&](size_t bytes)->char*{ char* p = w; w += (bytes + 255) & ~(size_t)255; return p; };
  float*  h     = (float*) alloc((size_t)N_TRAIN*ENC*4);
  float*  c     = (float*) alloc((size_t)N_TRAIN*ENC*4);
  __bf16* hbf   = (__bf16*)alloc((size_t)N_TRAIN*ENC*2);
  float*  z     = (float*) alloc((size_t)N_TRAIN*GATES*4);
  float*  pre   = (float*) alloc((size_t)N_ANTES*ENC*4);
  __bf16* St    = (__bf16*)alloc((size_t)N_ANTES*N_TRAIN*2);
  __bf16* Wabf  = (__bf16*)alloc((size_t)ENC*N_TRAIN*2);
  __bf16* ctxbf = (__bf16*)alloc((size_t)N_TRAIN*N_FEAT*2);
  __bf16* w1bf  = (__bf16*)alloc((size_t)ENC*N_FEAT*2);
  __bf16* w2bf  = (__bf16*)alloc((size_t)ENC*ENC*2);
  __bf16* whhbf = (__bf16*)alloc((size_t)GATES*ENC*2);
  __bf16* hidbf = (__bf16*)alloc((size_t)N_TRAIN*ENC*2);
  float*  sums  = (float*) alloc(GATES*4);
  float*  e     = (float*) alloc(ENC*4);
  float*  ewb   = (float*) alloc(ENC*4);
  float*  scores= (float*) alloc(N_ANTES*4);
  int*    idx   = (int*)   alloc(256);

  // ---- one-time precompute ----
  k_cvt<<<(N_TRAIN*N_FEAT+255)/256,256,0,stream>>>(context, ctxbf, N_TRAIN, N_FEAT, N_FEAT);
  k_cvt<<<(ENC*N_FEAT+255)/256,256,0,stream>>>(enc_w1, w1bf, ENC, N_FEAT, N_FEAT);
  k_cvt<<<(ENC*ENC+255)/256,256,0,stream>>>(enc_w2, w2bf, ENC, ENC, ENC);
  k_cvt<<<(GATES*ENC+255)/256,256,0,stream>>>(w_hh, whhbf, GATES, ENC, ENC);
  k_cvt<<<(ENC*N_TRAIN+255)/256,256,0,stream>>>(att_w1, Wabf, ENC, N_TRAIN, N_TRAIN+ENC);
  k_transpose_cvt<<<((size_t)N_TRAIN*N_ANTES+255)/256,256,0,stream>>>(S, St);
  k_rowsum<<<GATES/256,256,0,stream>>>(w_ih, sums);
  k_zero<<<(N_TRAIN*ENC+255)/256,256,0,stream>>>(c, N_TRAIN*ENC);

  // encoder L1: hid = relu(ctx @ W1^T + b1)  (f32 scratch in z)
  k_wmma_gemm<<<(N_TRAIN/16)*(ENC/64)/8,256,0,stream>>>(
      ctxbf, w1bf, enc_b1, z, hidbf, N_TRAIN, ENC, N_FEAT, N_FEAT, N_FEAT, ENC, 1);
  // encoder L2: h0 = hid @ W2^T + b2
  k_wmma_gemm<<<(N_TRAIN/16)*(ENC/64)/8,256,0,stream>>>(
      hidbf, w2bf, enc_b2, h, hbf, N_TRAIN, ENC, ENC, ENC, ENC, ENC, 0);
  // pre = S^T @ Wa^T   (2048x256, K=4096)
  k_wmma_gemm<<<(N_ANTES/16)*(ENC/64)/8,256,0,stream>>>(
      St, Wabf, nullptr, pre, nullptr, N_ANTES, ENC, N_TRAIN, N_TRAIN, N_TRAIN, ENC, 0);

  // ---- 16-step scan ----
  for (int step = 0; step < MAX_LEN; step++) {
    k_wmma_z<<<(N_TRAIN/128)*(GATES/64),256,64*ENC*2,stream>>>(
        hbf, whhbf, b_ih, b_hh, sums, S, idx, step == 0 ? 1 : 0, z);
    k_gates<<<N_TRAIN,256,0,stream>>>(z, c, h, hbf);
    k_col_mean<<<ENC,256,0,stream>>>(h, e);
    k_ewb<<<1,256,0,stream>>>(e, att_w1, att_b1, ewb);
    k_scores<<<N_ANTES/256,256,0,stream>>>(pre, ewb, att_w2, att_b2, scores);
    k_softmax<<<1,1024,0,stream>>>(scores, out + (size_t)step*N_ANTES, idx);
  }
}
